// Match_strategy_35493609734866
// MI455X (gfx1250) — matrix operand order repaired and checked by reading, not compiled, not run on hardware
//
#include <hip/hip_runtime.h>
#include <math.h>

// ---------------- types ----------------
typedef __attribute__((ext_vector_type(16))) __bf16 v16bf;
typedef __attribute__((ext_vector_type(8)))  float  v8f;

union Frag { uint4 q[2]; v16bf v; };

#define BATCH 16
#define NTOK  1024
#define DIM   768
#define MTOT  (BATCH * NTOK)   // 16384 rows

// ---------------- helpers ----------------
__device__ __forceinline__ unsigned short f2bf(float f) {
    unsigned int u = __float_as_uint(f);
    unsigned int r = u + 0x7FFFu + ((u >> 16) & 1u);   // RNE
    return (unsigned short)(r >> 16);
}
__device__ __forceinline__ float bf2f(unsigned short h) {
    return __uint_as_float(((unsigned int)h) << 16);
}
__device__ __forceinline__ float gelu_f(float v) {
    return 0.5f * v * (1.0f + erff(v * 0.70710678118654752f));
}
__device__ __forceinline__ float sigmoid_f(float t) {
    return 1.0f / (1.0f + expf(-t));
}
__device__ __forceinline__ v8f wmma_bf16(const Frag& a, const Frag& b, v8f c) {
    return __builtin_amdgcn_wmma_f32_16x16x32_bf16(false, a.v, false, b.v,
                                                   (short)0, c, false, false);
}

// ---------------- K0: weight transpose + bf16 convert: wt[o*K+k] = w[k*O+o] ----------------
__global__ void wtrans_kernel(const float* __restrict__ w, unsigned short* __restrict__ wt,
                              int K, int O) {
    int i = blockIdx.x * blockDim.x + threadIdx.x;
    if (i >= K * O) return;
    int o = i / K, k = i % K;
    wt[i] = f2bf(w[(size_t)k * O + o]);
}

// ---------------- K1: avg over tokens (two-stage: 8 token-chunks) ----------------
__global__ void avg_part_kernel(const float* __restrict__ x, float* __restrict__ pavg) {
    int i = blockIdx.x * blockDim.x + threadIdx.x;   // BATCH*DIM
    if (i >= BATCH * DIM) return;
    int b = i / DIM, d = i % DIM;
    int chunk = blockIdx.y;                           // 0..7, 128 tokens each
    const float* p = x + (size_t)b * NTOK * DIM + (size_t)chunk * 128 * DIM + d;
    float s = 0.f;
    #pragma unroll 4
    for (int n = 0; n < 128; ++n) s += p[(size_t)n * DIM];
    pavg[(size_t)chunk * BATCH * DIM + i] = s;
}
__global__ void avg_final_kernel(const float* __restrict__ pavg, float* __restrict__ avg) {
    int i = blockIdx.x * blockDim.x + threadIdx.x;
    if (i >= BATCH * DIM) return;
    float s = 0.f;
    #pragma unroll
    for (int c = 0; c < 8; ++c) s += pavg[(size_t)c * BATCH * DIM + i];
    avg[i] = s * (1.0f / (float)NTOK);
}

// ---------------- K2: per-row LN -> sigmoid -> A (bf16) + sq=||A||^2 ----------------
__global__ __launch_bounds__(256)
void rows_kernel(const float* __restrict__ x, const float* __restrict__ gamma,
                 const float* __restrict__ beta, const float* __restrict__ avg,
                 unsigned short* __restrict__ Abf, float* __restrict__ sq) {
    int row = blockIdx.x;                 // 0..MTOT-1
    int b = row >> 10;                    // /NTOK
    int tid = threadIdx.x, lane = tid & 31, wid = tid >> 5;
    const float* xr = x + (size_t)row * DIM;

    float v0 = xr[tid], v1 = xr[tid + 256], v2 = xr[tid + 512];
    float s = v0 + v1 + v2;
    float s2 = v0 * v0 + v1 * v1 + v2 * v2;
    __shared__ float red[2][8];
    #pragma unroll
    for (int m = 16; m >= 1; m >>= 1) {
        s  += __shfl_xor(s,  m, 32);
        s2 += __shfl_xor(s2, m, 32);
    }
    if (lane == 0) { red[0][wid] = s; red[1][wid] = s2; }
    __syncthreads();
    float ts = 0.f, ts2 = 0.f;
    #pragma unroll
    for (int i = 0; i < 8; ++i) { ts += red[0][i]; ts2 += red[1][i]; }
    float mu = ts * (1.0f / DIM);
    float var = ts2 * (1.0f / DIM) - mu * mu;
    float rstd = rsqrtf(var + 1e-5f);

    const float* avb = avg + (size_t)b * DIM;
    unsigned short* ar = Abf + (size_t)row * DIM;
    float asq = 0.f;
    #pragma unroll
    for (int e = 0; e < 3; ++e) {
        int d = tid + e * 256;
        float xv = (e == 0) ? v0 : (e == 1) ? v1 : v2;
        float ln = (xv - mu) * rstd * gamma[d] + beta[d];
        float a = avb[d] * sigmoid_f(ln);
        ar[d] = f2bf(a);
        asq += a * a;
    }
    #pragma unroll
    for (int m = 16; m >= 1; m >>= 1) asq += __shfl_xor(asq, m, 32);
    if (lane == 0) red[0][wid] = asq;
    __syncthreads();
    if (tid == 0) {
        float t = 0.f;
        #pragma unroll
        for (int i = 0; i < 8; ++i) t += red[0][i];
        sq[row] = t;
    }
}

// ---------------- K3: per-batch pairwise-distance row max via bf16 WMMA ----------------
// grid (NTOK/128, BATCH), block 256 (8 waves). Rows tile staged in LDS (padded).
// Each wave: 16 rows x 32 cols per iteration (2 accs, fa reused for 2 WMMAs).
#define SR_LD 776   // 768 + 8 pad (bank-conflict-free, 16B aligned rows)
__global__ __launch_bounds__(256)
void dist_kernel(const unsigned short* __restrict__ Abf, const float* __restrict__ sq,
                 float* __restrict__ mrow) {
    extern __shared__ unsigned short sRow[];   // 128 * SR_LD ushorts
    int b = blockIdx.y;
    int rowTile = blockIdx.x * 128;
    const unsigned short* Ab = Abf + (size_t)b * NTOK * DIM;
    const unsigned short* Arows = Ab + (size_t)rowTile * DIM;
    int tid = threadIdx.x;

    for (int c = tid; c < 128 * 96; c += 256) {     // 16B chunks: 96 per row
        int r = c / 96, o = (c % 96) * 8;
        *(uint4*)&sRow[r * SR_LD + o] = *(const uint4*)&Arows[(size_t)r * DIM + o];
    }
    __syncthreads();

    int w = tid >> 5, lane = tid & 31, ln = lane & 15, lh = lane >> 4;
    int rbase = w * 16;
    const float* sqb = sq + (size_t)b * NTOK;

    float sqm[8], rmax[8];
    #pragma unroll
    for (int r = 0; r < 8; ++r) {
        sqm[r] = sqb[rowTile + rbase + lh * 8 + r];
        rmax[r] = -1e30f;
    }

    const unsigned short* pa0 = &sRow[(rbase + ln) * SR_LD + (lh ? 16 : 0)];
    for (int jt = 0; jt < NTOK; jt += 32) {
        v8f acc0 = {0.f, 0.f, 0.f, 0.f, 0.f, 0.f, 0.f, 0.f};
        v8f acc1 = {0.f, 0.f, 0.f, 0.f, 0.f, 0.f, 0.f, 0.f};
        const unsigned short* pb0 = Ab + (size_t)(jt + ln) * DIM + lh * 16;
        const unsigned short* pb1 = pb0 + 16 * DIM;
        if (jt + 32 < NTOK) {   // prefetch next column panel (L2-resident)
            __builtin_prefetch(pb0 + 32 * DIM, 0, 1);
            __builtin_prefetch(pb1 + 32 * DIM, 0, 1);
        }
        #pragma unroll 4
        for (int k = 0; k < DIM; k += 32) {
            Frag fa, fb0, fb1;
            fa.q[0] = *(const uint4*)(pa0 + k);
            fa.q[1] = *(const uint4*)(pa0 + k + 8);
            fb0.q[0] = *(const uint4*)(pb0 + k);
            fb0.q[1] = *(const uint4*)(pb0 + k + 8);
            fb1.q[0] = *(const uint4*)(pb1 + k);
            fb1.q[1] = *(const uint4*)(pb1 + k + 8);
            acc0 = wmma_bf16(fa, fb0, acc0);
            acc1 = wmma_bf16(fa, fb1, acc1);
        }
        float sqn0 = sqb[jt + ln];
        float sqn1 = sqb[jt + 16 + ln];
        #pragma unroll
        for (int r = 0; r < 8; ++r) {
            float d20 = sqm[r] + sqn0 - 2.0f * acc0[r];
            float d21 = sqm[r] + sqn1 - 2.0f * acc1[r];
            rmax[r] = fmaxf(rmax[r], fmaxf(d20, d21));
        }
    }
    #pragma unroll
    for (int m = 8; m >= 1; m >>= 1) {
        #pragma unroll
        for (int r = 0; r < 8; ++r)
            rmax[r] = fmaxf(rmax[r], __shfl_xor(rmax[r], m, 32));
    }
    if (ln == 0) {
        #pragma unroll
        for (int r = 0; r < 8; ++r)
            mrow[(size_t)b * NTOK + rowTile + rbase + lh * 8 + r] =
                sqrtf(fmaxf(rmax[r], 1e-12f));
    }
}

// ---------------- K4: m_x = bf16(x + m_row) ----------------
__global__ void mx_kernel(const float* __restrict__ x, const float* __restrict__ mrow,
                          unsigned short* __restrict__ mx) {
    int i = blockIdx.x * blockDim.x + threadIdx.x;
    if (i >= MTOT * DIM) return;
    int row = i / DIM;
    mx[i] = f2bf(x[i] + mrow[row]);
}

// ---------------- K5..K8: double-buffered WMMA GEMM with fused epilogues ----------------
// C[m,o] = sum_k A[m,k] * Wt[o,k] ; block = 256 rows x 64 cols, 8 waves.
// Wave tile 32x64: 2 A-frags x 4 B-frags = 8 WMMAs per K-step; LDS ping-pong,
// global loads for step s+1 issued before the WMMAs of step s.
#define SA_LD 40
#define SW_LD 40
#define GROWS 256
__global__ __launch_bounds__(256)
void gemm_ep(const unsigned short* __restrict__ A0, const unsigned short* __restrict__ A1,
             const unsigned short* __restrict__ Wt, int ldw, int nphase,
             const float* __restrict__ bias, const float* __restrict__ xbuf,
             const unsigned short* __restrict__ m1buf,
             unsigned short* __restrict__ outb, float* __restrict__ outf, int mode) {
    __shared__ __align__(16) unsigned short sA[2][GROWS * SA_LD];   // 2 x 20 KB
    __shared__ __align__(16) unsigned short sW[2][64 * SW_LD];      // 2 x  5 KB
    int rowBase = blockIdx.x * GROWS;
    int oBase = blockIdx.y * 64;
    int tid = threadIdx.x, w = tid >> 5, lane = tid & 31;
    int ln = lane & 15, lh = lane >> 4;
    const int nsteps = nphase * (DIM / 32);

    v8f zero = {0.f, 0.f, 0.f, 0.f, 0.f, 0.f, 0.f, 0.f};
    v8f acc[2][4] = {{zero, zero, zero, zero}, {zero, zero, zero, zero}};

    uint4 ra[4], rw;
    auto loadRegs = [&](int s) {
        int p = s / (DIM / 32);
        int kk = (s % (DIM / 32)) * 32;
        const unsigned short* Asrc = p ? A1 : A0;
        int ko = p * DIM;
        #pragma unroll
        for (int it = 0; it < 4; ++it) {
            int c = tid + it * 256;
            int r = c >> 2, k2 = (c & 3) * 8;
            ra[it] = *(const uint4*)&Asrc[(size_t)(rowBase + r) * DIM + kk + k2];
        }
        int r = tid >> 2, k2 = (tid & 3) * 8;
        rw = *(const uint4*)&Wt[(size_t)(oBase + r) * ldw + ko + kk + k2];
    };
    auto storeRegs = [&](int buf) {
        #pragma unroll
        for (int it = 0; it < 4; ++it) {
            int c = tid + it * 256;
            int r = c >> 2, k2 = (c & 3) * 8;
            *(uint4*)&sA[buf][r * SA_LD + k2] = ra[it];
        }
        int r = tid >> 2, k2 = (tid & 3) * 8;
        *(uint4*)&sW[buf][r * SW_LD + k2] = rw;
    };

    loadRegs(0);
    storeRegs(0);
    __syncthreads();
    int cur = 0;
    for (int s = 0; s < nsteps; ++s) {
        bool hasNext = (s + 1 < nsteps);
        if (hasNext) loadRegs(s + 1);            // global loads in flight during WMMAs

        Frag fa0, fa1;
        const unsigned short* pa = &sA[cur][(w * 32 + ln) * SA_LD + (lh ? 16 : 0)];
        fa0.q[0] = *(const uint4*)pa;
        fa0.q[1] = *(const uint4*)(pa + 8);
        const unsigned short* pa1 = pa + 16 * SA_LD;
        fa1.q[0] = *(const uint4*)pa1;
        fa1.q[1] = *(const uint4*)(pa1 + 8);
        #pragma unroll
        for (int t = 0; t < 4; ++t) {
            Frag fb;
            const unsigned short* pb = &sW[cur][(t * 16 + ln) * SW_LD + lh * 16];
            fb.q[0] = *(const uint4*)pb;
            fb.q[1] = *(const uint4*)(pb + 8);
            acc[0][t] = wmma_bf16(fa0, fb, acc[0][t]);
            acc[1][t] = wmma_bf16(fa1, fb, acc[1][t]);
        }

        if (hasNext) {
            storeRegs(cur ^ 1);                  // other buffer: no conflict with readers
            __syncthreads();                     // single barrier per K-step
            cur ^= 1;
        }
    }

    // epilogue
    #pragma unroll
    for (int rt = 0; rt < 2; ++rt) {
        #pragma unroll
        for (int t = 0; t < 4; ++t) {
            int o = oBase + t * 16 + ln;
            float bz = bias[o];
            #pragma unroll
            for (int r = 0; r < 8; ++r) {
                int m = rowBase + w * 32 + rt * 16 + lh * 8 + r;
                size_t idx = (size_t)m * DIM + o;
                float v = acc[rt][t][r] + bz;
                if (mode == 1) {
                    outb[idx] = f2bf(gelu_f(v));                          // m1
                } else if (mode == 2) {
                    float g = gelu_f(v);                                  // W
                    outb[idx] = f2bf(xbuf[idx] * bf2f(m1buf[idx]) * g);   // m1*x*W
                } else if (mode == 3) {
                    outb[idx] = f2bf(xbuf[idx] * gelu_f(v));              // x*l3
                } else {
                    outf[idx] = gelu_f(v);                                // final
                }
            }
        }
    }
}

// ---------------- launch ----------------
extern "C" void kernel_launch(void* const* d_in, const int* in_sizes, int n_in,
                              void* d_out, int out_size, void* d_ws, size_t ws_size,
                              hipStream_t stream) {
    (void)in_sizes; (void)n_in; (void)out_size; (void)ws_size;
    const float* x     = (const float*)d_in[0];
    const float* gamma = (const float*)d_in[1];
    const float* beta  = (const float*)d_in[2];
    const float* w1    = (const float*)d_in[3];
    const float* b1    = (const float*)d_in[4];
    const float* w2    = (const float*)d_in[5];
    const float* b2    = (const float*)d_in[6];
    const float* w3    = (const float*)d_in[7];
    const float* b3    = (const float*)d_in[8];
    const float* w4    = (const float*)d_in[9];
    const float* b4    = (const float*)d_in[10];
    float* out = (float*)d_out;

    // workspace layout (all 256B aligned)
    char* ws = (char*)d_ws;
    const size_t SZ_AVG = (size_t)BATCH * DIM * 4;          // 49152
    const size_t SZ_ROW = (size_t)BATCH * NTOK * 4;         // 65536
    const size_t SZ_ACT = (size_t)MTOT * DIM * 2;           // 25165824 (bf16)
    size_t off = 0;
    float* avg  = (float*)(ws + off); off += SZ_AVG;
    float* pavg = (float*)(ws + off); off += 8 * SZ_AVG;    // 8 partial chunks
    float* sq   = (float*)(ws + off); off += SZ_ROW;
    float* mrow = (float*)(ws + off); off += SZ_ROW;
    unsigned short* Abf  = (unsigned short*)(ws + off); off += SZ_ACT;
    unsigned short* mxbf = (unsigned short*)(ws + off); off += SZ_ACT;
    unsigned short* m1bf = (unsigned short*)(ws + off); off += SZ_ACT;
    unsigned short* h3bf = (unsigned short*)(ws + off); off += SZ_ACT;
    unsigned short* xl3  = (unsigned short*)(ws + off); off += SZ_ACT;
    unsigned short* w1t  = (unsigned short*)(ws + off); off += (size_t)DIM * 2 * DIM * 2;
    unsigned short* w2t  = (unsigned short*)(ws + off); off += (size_t)DIM * DIM * 2;
    unsigned short* w3t  = (unsigned short*)(ws + off); off += (size_t)DIM * DIM * 2;
    unsigned short* w4t  = (unsigned short*)(ws + off); off += (size_t)DIM * DIM * 2;

    // weights -> bf16 transposed [O][K]
    wtrans_kernel<<<(2 * DIM * DIM + 255) / 256, 256, 0, stream>>>(w1, w1t, 2 * DIM, DIM);
    wtrans_kernel<<<(DIM * DIM + 255) / 256, 256, 0, stream>>>(w2, w2t, DIM, DIM);
    wtrans_kernel<<<(DIM * DIM + 255) / 256, 256, 0, stream>>>(w3, w3t, DIM, DIM);
    wtrans_kernel<<<(DIM * DIM + 255) / 256, 256, 0, stream>>>(w4, w4t, DIM, DIM);

    // avg_x (two-stage for parallelism: 384 blocks then 48)
    avg_part_kernel<<<dim3((BATCH * DIM + 255) / 256, 8), 256, 0, stream>>>(x, pavg);
    avg_final_kernel<<<(BATCH * DIM + 255) / 256, 256, 0, stream>>>(pavg, avg);
    // A + sq
    rows_kernel<<<MTOT, 256, 0, stream>>>(x, gamma, beta, avg, Abf, sq);
    // pairwise distance row-max (194KB dynamic LDS of CDNA5's 320KB)
    dist_kernel<<<dim3(NTOK / 128, BATCH), 256, 128 * SR_LD * 2, stream>>>(Abf, sq, mrow);
    // m_x
    mx_kernel<<<(MTOT * DIM + 255) / 256, 256, 0, stream>>>(x, mrow, mxbf);

    dim3 ggrid(MTOT / GROWS, DIM / 64);
    // layer1: m1 = gelu([A, m_x] @ w1 + b1)   (two K-phases over w1t, ldw=1536)
    gemm_ep<<<ggrid, 256, 0, stream>>>(Abf, mxbf, w1t, 2 * DIM, 2, b1,
                                       nullptr, nullptr, m1bf, nullptr, 1);
    // layer2 fused: h3 = m1 * x * gelu(m1 @ w2 + b2)
    gemm_ep<<<ggrid, 256, 0, stream>>>(m1bf, nullptr, w2t, DIM, 1, b2,
                                       x, m1bf, h3bf, nullptr, 2);
    // layer3 fused: xl3 = x * gelu(h3 @ w3 + b3)
    gemm_ep<<<ggrid, 256, 0, stream>>>(h3bf, nullptr, w3t, DIM, 1, b3,
                                       x, nullptr, xl3, nullptr, 3);
    // layer4: out = gelu(xl3 @ w4 + b4)
    gemm_ep<<<ggrid, 256, 0, stream>>>(xl3, nullptr, w4t, DIM, 1, b4,
                                       nullptr, nullptr, nullptr, out, 4);
}
